// GatedConvSSM2D_V2_19628000543392
// MI455X (gfx1250) — matrix operand order repaired and checked by reading, not compiled, not run on hardware
//
#include <hip/hip_runtime.h>
#include <hip/hip_bf16.h>

typedef __attribute__((ext_vector_type(16))) _Float16 v16h;
typedef __attribute__((ext_vector_type(8)))  _Float16 v8h;
typedef __attribute__((ext_vector_type(8)))  float    v8f;

#define BATCH 4
#define HH 128
#define WW 128
#define CC 128
#define KS 7
#define MTOT (BATCH * HH * WW)   // 65536 GEMM rows

// ---------------------------------------------------------------------------
// Prep 1: transpose a (C,7,7) kernel to (7*7, C) so the conv pass reads it
// coalesced along threadIdx.x == c.
__global__ void transpose_k7(const float* __restrict__ src, float* __restrict__ dst) {
    int tid = blockIdx.x * blockDim.x + threadIdx.x;
    if (tid < KS * KS * CC) {
        int t = tid / CC, c = tid % CC;            // t = i*7+j
        dst[t * CC + c] = src[c * KS * KS + t];
    }
}

// Prep 2: convert W_out (C,C) f32 -> f16 for the WMMA GEMM.
__global__ void cvt_w16(const float* __restrict__ W, _Float16* __restrict__ w16) {
    int tid = blockIdx.x * blockDim.x + threadIdx.x;
    if (tid < CC * CC) w16[tid] = (_Float16)W[tid];
}

// ---------------------------------------------------------------------------
// Fused pass: 3 depthwise circular 7x7 convs + a_spatial + gates + closed-form
// T-step scan:  h_final = (softplus(b_t)+eps) * (1 - sigmoid(-a_t)^T)
// One block (128 threads == channels) per pixel; x taps are C-contiguous so
// every tap load is a fully coalesced 512B row, and all taps stay L2-resident
// (whole x is 33.5 MB << 192 MB L2).
__global__ void __launch_bounds__(CC)
conv_gate_scan(const float* __restrict__ x,
               const float* __restrict__ kAt,  const float* __restrict__ kBt,
               const float* __restrict__ kGAt, const float* __restrict__ kGBt,
               const int*   __restrict__ Tptr,
               _Float16* __restrict__ h16) {
    const int c   = threadIdx.x;
    const int pix = blockIdx.x;                 // 0 .. B*H*W-1  (== GEMM row m)
    const int w   = pix % WW;
    const int hw  = pix / WW;
    const int h   = hw % HH;
    const int b   = hw / HH;

    float accB = 0.f, accGA = 0.f, accGB = 0.f;
#pragma unroll
    for (int i = 0; i < KS; ++i) {
        int hh = h - i + 3;
        hh += (hh < 0) ? HH : 0;
        hh -= (hh >= HH) ? HH : 0;
        const float* xrow = x + ((size_t)(b * HH + hh) * WW) * CC;
        // hint the tap row toward the caches (global_prefetch_b8)
        __builtin_prefetch(xrow + (size_t)w * CC + c, 0, 1);
#pragma unroll
        for (int j = 0; j < KS; ++j) {
            int ww = w - j + 3;
            ww += (ww < 0) ? WW : 0;
            ww -= (ww >= WW) ? WW : 0;
            float xv = xrow[(size_t)ww * CC + c];
            int   t  = i * KS + j;
            accB  = fmaf(xv, kBt [t * CC + c], accB);
            accGA = fmaf(xv, kGAt[t * CC + c], accGA);
            accGB = fmaf(xv, kGBt[t * CC + c], accGB);
        }
    }

    // a_spatial = padded A kernel: nonzero iff (h+3)%H < 7 and (w+3)%W < 7
    int ai = h + 3; ai -= (ai >= HH) ? HH : 0;
    int aj = w + 3; aj -= (aj >= WW) ? WW : 0;
    float a_sp = (ai < KS && aj < KS) ? kAt[(ai * KS + aj) * CC + c] : 0.f;

    float gA  = 1.f / (1.f + __expf(-accGA));
    float gB  = 1.f / (1.f + __expf(-accGB));
    float a_t = a_sp * gA;
    float b_t = accB * gB;

    // stable softplus
    float sp = fmaxf(b_t, 0.f) + log1pf(__expf(-fabsf(b_t)));
    float cc = 1.f / (1.f + __expf(a_t));       // sigmoid(-a_t) in (0,1)
    int   T  = *Tptr;
    float cT = 1.f;
    for (int t = 0; t < T; ++t) cT *= cc;       // c^T (T==8)
    float hf = (sp + 1e-6f) * (1.f - cT);

    h16[(size_t)pix * CC + c] = (_Float16)hf;
}

// ---------------------------------------------------------------------------
// WMMA GEMM + bias: out[m,n] = sum_k h16[m,k] * W[k,n] + b_out[n]
// M=65536, K=N=128.
//
// Block = 128 threads (4 waves).  Each wave owns 4 stacked 16x16 M-tiles of
// one N-tile.  The W column tile is staged in LDS *pre-swizzled into
// fragment-major order* ldsB[ks][lane][e], so each lane's v16h B fragment is
// one contiguous 32-byte chunk -> two ds_load_b128, no half-packing moves.
// The 4 B fragments (K=128) are hoisted into registers and reused across the
// 4 M-tiles -> 16 v_wmma per wave.
//
// A fragment (ISA 7.12.2, 16-bit A 16x32): lane L holds row M=L%16; halves
// e=0..7 are K=kbase..kbase+7 and e=8..15 are K=kbase+16..kbase+23 with
// kbase = (L<16)?0:8 -> two contiguous 16B global loads + shufflevector.
__global__ void __launch_bounds__(128)
wmma_gemm_bias(const _Float16* __restrict__ A,    // (M,128) f16
               const _Float16* __restrict__ Wm,   // (128,128) f16
               const float*    __restrict__ bias, // (128,) f32
               float*          __restrict__ out)  // (M,128) f32
{
    __shared__ _Float16 ldsB[4 * 32 * 16];        // [ks][lane][e], 4 KB

    const int n0 = blockIdx.y * 16;

    // Stage + swizzle: ldsB[((ks*32+lane)*16)+e] = W[ks*32 + kbaseB(lane) + (e<8 ? e : e+8)][n0 + (lane&15)], kbaseB = (lane<16)?0:8
    for (int f = threadIdx.x; f < 4 * 32 * 16; f += blockDim.x) {
        int e    = f & 15;
        int ln   = (f >> 4) & 31;
        int ks   = f >> 9;
        int k    = ks * 32 + ((ln & 16) ? 8 : 0) + ((e < 8) ? e : e + 8);
        int ncol = n0 + (ln & 15);
        ldsB[f] = Wm[k * CC + ncol];
    }
    __syncthreads();

    const int wave   = threadIdx.x >> 5;
    const int lane   = threadIdx.x & 31;
    const int nn     = lane & 15;
    const int kbaseA = (lane < 16) ? 0 : 8;
    const int m0     = blockIdx.x * 256 + wave * 64;   // 4 tiles of 16 rows

    // Hoist all 4 B fragments (full K) into registers: 2x ds_load_b128 each.
    v16h bfr[4];
#pragma unroll
    for (int ks = 0; ks < 4; ++ks)
        bfr[ks] = *(const v16h*)&ldsB[(ks * 32 + lane) * 16];

    const float bv = bias[n0 + nn];

#pragma unroll
    for (int tile = 0; tile < 4; ++tile) {
        const int mrow = m0 + tile * 16 + nn;          // this lane's A row
        const _Float16* arow = A + (size_t)mrow * CC;

        v8f acc;
#pragma unroll
        for (int r = 0; r < 8; ++r) acc[r] = bv;       // bias is per-column

#pragma unroll
        for (int ks = 0; ks < 4; ++ks) {
            const int k0 = ks * 32 + kbaseA;
            v8h lo = *(const v8h*)(arow + k0);         // e=0..7  : K=k0..k0+7
            v8h hi = *(const v8h*)(arow + k0 + 16);    // e=8..15 : K=k0+16..k0+23
            v16h afr = __builtin_shufflevector(lo, hi,
                0, 1, 2, 3, 4, 5, 6, 7, 8, 9, 10, 11, 12, 13, 14, 15);
            acc = __builtin_amdgcn_wmma_f32_16x16x32_f16(
                      /*neg_a=*/false, afr, /*neg_b=*/false, bfr[ks],
                      /*c_mod=*/(short)0, acc, /*reuse_a=*/false, /*reuse_b=*/false);
        }

        const int mtop = m0 + tile * 16 + ((lane < 16) ? 0 : 8);
#pragma unroll
        for (int r = 0; r < 8; ++r)
            out[(size_t)(mtop + r) * CC + n0 + nn] = acc[r];
    }
}

// ---------------------------------------------------------------------------
extern "C" void kernel_launch(void* const* d_in, const int* in_sizes, int n_in,
                              void* d_out, int out_size, void* d_ws, size_t ws_size,
                              hipStream_t stream) {
    const float* x    = (const float*)d_in[0];   // (4,128,128,128)
    const float* Ak   = (const float*)d_in[1];   // (128,7,7)
    const float* Bk   = (const float*)d_in[2];
    const float* GAk  = (const float*)d_in[3];
    const float* GBk  = (const float*)d_in[4];
    const float* Wout = (const float*)d_in[5];   // (128,128)
    const float* bout = (const float*)d_in[6];   // (128,)
    const int*   Tptr = (const int*)d_in[7];     // scalar T

    // Workspace layout (all offsets 256B-aligned):
    //   h16 : M*K f16     = 16,777,216 B
    //   w16 : 128*128 f16 =     32,768 B
    //   4x transposed (49,128) f32 kernels = 4 * 25,088 B
    char* ws = (char*)d_ws;
    _Float16* h16 = (_Float16*)(ws);
    _Float16* w16 = (_Float16*)(ws + 16777216);
    float*    kAt  = (float*)(ws + 16777216 + 32768);
    float*    kBt  = (float*)(ws + 16777216 + 32768 + 25088);
    float*    kGAt = (float*)(ws + 16777216 + 32768 + 2 * 25088);
    float*    kGBt = (float*)(ws + 16777216 + 32768 + 3 * 25088);

    // Prep: kernel transposes + W_out f16 conversion
    const int ktElems = KS * KS * CC;            // 6272
    transpose_k7<<<(ktElems + 255) / 256, 256, 0, stream>>>(Ak,  kAt);
    transpose_k7<<<(ktElems + 255) / 256, 256, 0, stream>>>(Bk,  kBt);
    transpose_k7<<<(ktElems + 255) / 256, 256, 0, stream>>>(GAk, kGAt);
    transpose_k7<<<(ktElems + 255) / 256, 256, 0, stream>>>(GBk, kGBt);
    cvt_w16<<<(CC * CC + 255) / 256, 256, 0, stream>>>(Wout, w16);

    // Fused conv + gates + closed-form scan -> f16 GEMM operand
    conv_gate_scan<<<MTOT, CC, 0, stream>>>(x, kAt, kBt, kGAt, kGBt, Tptr, h16);

    // WMMA channel-mixing GEMM + bias -> output (f32)
    // Each block: 4 waves x 4 M-tiles x 16 rows = 256 rows of one N-tile.
    dim3 ggrid(MTOT / 256, CC / 16);             // (256, 8)
    wmma_gemm_bias<<<ggrid, 128, 0, stream>>>(h16, w16, bout, (float*)d_out);
}
